// ResidualVQLayer_4440996184535
// MI455X (gfx1250) — matrix operand order repaired and checked by reading, not compiled, not run on hardware
//
#include <hip/hip_runtime.h>
#include <cstdint>
#include <cstddef>

// ---------------- problem constants ----------------
#define N_TOK    131072
#define DIM      512
#define K_CODES  1024
#define TILE_TOK 128                 // tokens per workgroup (8 waves x 16)
#define CHUNK    32                  // codes per LDS chunk
#define NCHUNK   (K_CODES / CHUNK)   // 32
#define PITCH    520                 // halves per LDS codebook row (row stride 1040B = 4-bank skew, 16B aligned)
#define BUF_BYTES (CHUNK * PITCH * 2)        // 33,280 B per buffer
#define DYN_LDS_BYTES (2 * BUF_BYTES)        // 66,560 B (double buffer)

typedef __bf16 v16bf __attribute__((ext_vector_type(16)));
typedef __bf16 v2bf  __attribute__((ext_vector_type(2)));
typedef float  v8f   __attribute__((ext_vector_type(8)));

union BFrag { v16bf v; uint4 q[2]; unsigned u[8]; };

__device__ __forceinline__ unsigned short f2bf(float f) {
  union { float f; unsigned u; } c; c.f = f;
  unsigned r = c.u + 0x7FFFu + ((c.u >> 16) & 1u);   // round-to-nearest-even
  return (unsigned short)(r >> 16);
}
__device__ __forceinline__ unsigned pk2(float lo, float hi) {
#if __has_builtin(__builtin_amdgcn_cvt_pk_bf16_f32)
  union { v2bf v; unsigned u; } c;
  c.v = __builtin_amdgcn_cvt_pk_bf16_f32(lo, hi);
  return c.u;
#else
  return (unsigned)f2bf(lo) | ((unsigned)f2bf(hi) << 16);
#endif
}

// low 32 bits of a generic pointer to a __shared__ object = LDS byte address
__device__ __forceinline__ unsigned lds_addr_of(const void* p) {
  return (unsigned)(unsigned long long)p;
}
// CDNA5 async L2->LDS copy, 16B, GVS mode (saddr base + 32-bit voffset); ASYNCcnt-tracked
__device__ __forceinline__ void async_ld16(unsigned lds_addr, const void* gbase, unsigned byte_off) {
  asm volatile("global_load_async_to_lds_b128 %0, %1, %2"
               :: "v"(lds_addr), "v"(byte_off), "s"(gbase)
               : "memory");
}
__device__ __forceinline__ void wait_async0() {
  asm volatile("s_wait_asynccnt 0x0" ::: "memory");
}

// ---------------- kernel 1: codebook f32 -> bf16 + ||e||^2 ----------------
__global__ __launch_bounds__(256)
void vq_prep(const float* __restrict__ emb, unsigned short* __restrict__ cbg,
             float* __restrict__ enorm) {
  __shared__ float red[256];
  const int k = blockIdx.x, tid = threadIdx.x;
  float2 v = *(const float2*)(emb + (size_t)k * DIM + 2 * tid);
  *(unsigned*)(cbg + (size_t)k * DIM + 2 * tid) = pk2(v.x, v.y);
  red[tid] = v.x * v.x + v.y * v.y;
  __syncthreads();
  for (int s = 128; s > 0; s >>= 1) { if (tid < s) red[tid] += red[tid + s]; __syncthreads(); }
  if (tid == 0) enorm[k] = red[0];
}

// ---------------- kernel 2: WMMA distances + argmin + fused epilogue ----------------
__global__ __launch_bounds__(256, 1)
void vq_main(const float* __restrict__ z, const float* __restrict__ emb,
             const unsigned short* __restrict__ cbg, const float* __restrict__ enorm,
             float* __restrict__ o_zq, float* __restrict__ o_res, float* __restrict__ o_idx,
             float* __restrict__ o_loss, float* __restrict__ o_cnt, float* __restrict__ o_dw) {
  extern __shared__ __align__(16) unsigned short cbuf_dyn[];   // 2 x (CHUNK x PITCH) bf16
  __shared__ int   idx_lds[TILE_TOK];
  __shared__ float red[256];

  const int tid     = threadIdx.x;
  const int wv      = tid >> 5;
  const int lane    = tid & 31;
  const int ln16    = lane & 15;
  const int halfsel = lane >> 4;          // 0: lanes 0-15, 1: lanes 16-31
  const int t0      = blockIdx.x * TILE_TOK;

  const unsigned lbase0 = lds_addr_of(&cbuf_dyn[0]);
  const unsigned lbase1 = lbase0 + BUF_BYTES;

  // ---- A fragments in registers: 16 tokens x 512 dims bf16 per wave (128 VGPRs).
  // A 16x32 layout (matches B): lane<16 holds K 0..15, lane>=16 holds K 16..31.
  const int row = t0 + wv * 16 + ln16;
  const float* zr = z + (size_t)row * DIM;
  BFrag A[16];
#pragma unroll
  for (int ks = 0; ks < 16; ++ks) {
    const int k0 = ks * 32 + halfsel * 16;
    float4 x0 = *(const float4*)(zr + k0);
    float4 x1 = *(const float4*)(zr + k0 + 4);
    float4 y0 = *(const float4*)(zr + k0 + 8);
    float4 y1 = *(const float4*)(zr + k0 + 12);
    A[ks].u[0] = pk2(x0.x, x0.y); A[ks].u[1] = pk2(x0.z, x0.w);
    A[ks].u[2] = pk2(x1.x, x1.y); A[ks].u[3] = pk2(x1.z, x1.w);
    A[ks].u[4] = pk2(y0.x, y0.y); A[ks].u[5] = pk2(y0.z, y0.w);
    A[ks].u[6] = pk2(y1.x, y1.y); A[ks].u[7] = pk2(y1.z, y1.w);
  }

  // ---- async-stage chunk 0 (32 codes x 512 bf16 = 32KB), padded pitch
  {
    const char* gsrc = (const char*)cbg;   // chunk 0
#pragma unroll
    for (int i = 0; i < 8; ++i) {
      int u = tid + i * 256;               // 16B unit; 64 units per code row
      int rr = u >> 6, col = u & 63;
      async_ld16(lbase0 + (unsigned)rr * (PITCH * 2) + col * 16, gsrc, (unsigned)u * 16);
    }
  }
  wait_async0();
  __syncthreads();

  float best[8]; int bidx[8];
#pragma unroll
  for (int r = 0; r < 8; ++r) { best[r] = 3.4e38f; bidx[r] = 0; }

  for (int c = 0; c < NCHUNK; ++c) {
    const int cur = c & 1;
    // kick off async staging of next chunk into the idle buffer (no VGPR cost)
    if (c + 1 < NCHUNK) {
      const char* gsrc = (const char*)(cbg + (size_t)(c + 1) * CHUNK * DIM);
      const unsigned nbase = cur ? lbase0 : lbase1;
#pragma unroll
      for (int i = 0; i < 8; ++i) {
        int u = tid + i * 256;
        int rr = u >> 6, col = u & 63;
        async_ld16(nbase + (unsigned)rr * (PITCH * 2) + col * 16, gsrc, (unsigned)u * 16);
      }
    }

    const float en0 = enorm[c * CHUNK + ln16];
    const float en1 = enorm[c * CHUNK + 16 + ln16];
    const char* cb_cur = (const char*)cbuf_dyn + (size_t)cur * BUF_BYTES;

    v8f acc0 = {0.f, 0.f, 0.f, 0.f, 0.f, 0.f, 0.f, 0.f};
    v8f acc1 = acc0;
#pragma unroll
    for (int ks = 0; ks < 16; ++ks) {
      // B 32x16 layout: col N = lane&15; lane<16 holds K 0-15, lane>=16 holds K 16-31
      const char* bbase = cb_cur + (size_t)(ks * 64 + halfsel * 32);
      BFrag b0, b1;
      const char* p0 = bbase + (size_t)ln16 * (PITCH * 2);
      b0.q[0] = *(const uint4*)p0;  b0.q[1] = *(const uint4*)(p0 + 16);
      const char* p1 = bbase + (size_t)(16 + ln16) * (PITCH * 2);
      b1.q[0] = *(const uint4*)p1;  b1.q[1] = *(const uint4*)(p1 + 16);
      acc0 = __builtin_amdgcn_wmma_f32_16x16x32_bf16(false, A[ks].v, false, b0.v,
                                                     (short)0, acc0, false, false);
      acc1 = __builtin_amdgcn_wmma_f32_16x16x32_bf16(false, A[ks].v, false, b1.v,
                                                     (short)0, acc1, false, false);
    }
    // running argmin of dist = ||e||^2 - 2 z.e  (||z||^2 constant per row)
    const int cg0 = c * CHUNK + ln16;
    const int cg1 = cg0 + 16;
#pragma unroll
    for (int r = 0; r < 8; ++r) {
      float d0 = en0 - 2.f * acc0[r];
      float d1 = en1 - 2.f * acc1[r];
      if (d0 < best[r]) { best[r] = d0; bidx[r] = cg0; }
      if (d1 < best[r]) { best[r] = d1; bidx[r] = cg1; }
    }

    if (c + 1 < NCHUNK) wait_async0();   // next chunk fully in LDS
    __syncthreads();                     // everyone done with cur, next visible to all
  }

  // cross-lane argmin within each 16-lane half (D tile: vgpr r -> row r / r+8)
#pragma unroll
  for (int r = 0; r < 8; ++r) {
    float v = best[r]; int id = bidx[r];
#pragma unroll
    for (int m = 8; m >= 1; m >>= 1) {
      float ov = __shfl_xor(v, m, 32);
      int   oi = __shfl_xor(id, m, 32);
      if (ov < v || (ov == v && oi < id)) { v = ov; id = oi; }  // first-index tiebreak
    }
    if (ln16 == r) {
      int tl = wv * 16 + r + 8 * halfsel;
      idx_lds[tl] = id;
      o_idx[t0 + tl] = (float)id;
    }
  }
  __syncthreads();

  // ---- fused epilogue: z_q_st, residual, loss, dw/count accumulation
  float lsum = 0.f;
  for (int tl = 0; tl < TILE_TOK; ++tl) {
    const int id = idx_lds[tl];
    const size_t zoff = (size_t)(t0 + tl) * DIM + 2 * tid;
    float2 zv = *(const float2*)(z + zoff);
    float2 ev = *(const float2*)(emb + (size_t)id * DIM + 2 * tid);
    *(float2*)(o_zq + zoff) = ev;                       // z_q_st == z_q numerically
    float2 rv; rv.x = zv.x - ev.x; rv.y = zv.y - ev.y;
    *(float2*)(o_res + zoff) = rv;
    lsum += rv.x * rv.x + rv.y * rv.y;
    atomicAdd(o_dw + (size_t)id * DIM + 2 * tid,     zv.x);
    atomicAdd(o_dw + (size_t)id * DIM + 2 * tid + 1, zv.y);
    if (tid == 0) atomicAdd(o_cnt + id, 1.f);
  }
  red[tid] = lsum; __syncthreads();
  for (int s = 128; s > 0; s >>= 1) { if (tid < s) red[tid] += red[tid + s]; __syncthreads(); }
  if (tid == 0) atomicAdd(o_loss, red[0]);
}

// ---------------- kernel 3: cluster-size EMA + normalize, perplexity, loss scale ----------------
__global__ __launch_bounds__(1024)
void vq_finalize(const float* __restrict__ ema_cs, float* __restrict__ o_loss,
                 float* __restrict__ o_perp, float* __restrict__ o_nc) {
  __shared__ float red[1024];
  const int k = threadIdx.x;
  const float cnt = o_nc[k];                           // raw counts accumulated by vq_main
  const float nc  = 0.99f * ema_cs[k] + 0.01f * cnt;
  red[k] = nc; __syncthreads();
  for (int s = 512; s > 0; s >>= 1) { if (k < s) red[k] += red[k + s]; __syncthreads(); }
  const float ntot = red[0]; __syncthreads();
  const float avg = cnt * (1.f / (float)N_TOK);
  red[k] = avg * logf(avg + 1e-10f); __syncthreads();
  for (int s = 512; s > 0; s >>= 1) { if (k < s) red[k] += red[k + s]; __syncthreads(); }
  const float psum = red[0];
  o_nc[k] = (nc + 1e-5f) / (ntot + (float)K_CODES * 1e-5f) * ntot;
  if (k == 0) {
    o_perp[0] = expf(-psum);
    o_loss[0] = 0.25f * o_loss[0] / ((float)N_TOK * (float)DIM);
  }
}

// ---------------- kernel 4: new_ema_embedding & new_embedding_weight ----------------
__global__ __launch_bounds__(256)
void vq_update(const float* __restrict__ ema_emb, const float* __restrict__ nc,
               float* __restrict__ o_ema, float* __restrict__ o_w) {
  const size_t i = ((size_t)blockIdx.x * 256 + threadIdx.x) * 4;
  const int k = (int)(i >> 9);                         // / DIM
  float4 dw = *(const float4*)(o_ema + i);             // raw dw accumulated by vq_main
  float4 em = *(const float4*)(ema_emb + i);
  float4 ne;
  ne.x = 0.99f * em.x + 0.01f * dw.x;
  ne.y = 0.99f * em.y + 0.01f * dw.y;
  ne.z = 0.99f * em.z + 0.01f * dw.z;
  ne.w = 0.99f * em.w + 0.01f * dw.w;
  *(float4*)(o_ema + i) = ne;
  const float inv = 1.f / nc[k];
  float4 w; w.x = ne.x * inv; w.y = ne.y * inv; w.z = ne.z * inv; w.w = ne.w * inv;
  *(float4*)(o_w + i) = w;
}

// ---------------- host side ----------------
extern "C" void kernel_launch(void* const* d_in, const int* in_sizes, int n_in,
                              void* d_out, int out_size, void* d_ws, size_t ws_size,
                              hipStream_t stream) {
  (void)in_sizes; (void)n_in; (void)out_size; (void)ws_size;
  const float* z       = (const float*)d_in[0];
  const float* emb     = (const float*)d_in[1];
  const float* ema_cs  = (const float*)d_in[2];
  const float* ema_emb = (const float*)d_in[3];

  float* out = (float*)d_out;
  const size_t ND = (size_t)N_TOK * DIM;
  const size_t KD = (size_t)K_CODES * DIM;
  float* o_zq   = out;                 // [N,D]
  float* o_res  = out + ND;            // [N,D]
  float* o_loss = out + 2 * ND;        // scalar
  float* o_perp = o_loss + 1;          // scalar
  float* o_idx  = o_perp + 1;          // [N] (as float)
  float* o_w    = o_idx + N_TOK;       // [K,D] new_embedding_weight
  float* o_nc   = o_w + KD;            // [K]   new_cluster_size (holds raw counts first)
  float* o_ema  = o_nc + K_CODES;      // [K,D] new_ema_embedding (holds raw dw first)

  unsigned short* cbg = (unsigned short*)d_ws;                 // bf16 codebook, 1 MB
  float* enorm = (float*)((char*)d_ws + KD * 2);               // ||e||^2, 4 KB

  hipMemsetAsync(o_loss, 0, sizeof(float), stream);
  hipMemsetAsync(o_nc,   0, K_CODES * sizeof(float), stream);
  hipMemsetAsync(o_ema,  0, KD * sizeof(float), stream);

  vq_prep<<<K_CODES, 256, 0, stream>>>(emb, cbg, enorm);
  vq_main<<<N_TOK / TILE_TOK, 256, DYN_LDS_BYTES, stream>>>(z, emb, cbg, enorm,
                                                            o_zq, o_res, o_idx, o_loss, o_nc, o_ema);
  vq_finalize<<<1, 1024, 0, stream>>>(ema_cs, o_loss, o_perp, o_nc);
  vq_update<<<(unsigned)(KD / 1024), 256, 0, stream>>>(ema_emb, o_nc, o_ema, o_w);
}
